// BiLSTM_66563403153516
// MI455X (gfx1250) — compile-verified, matrix operand order checked by reading, not run on hardware
//
#include <hip/hip_runtime.h>
#include <math.h>

// ---------------------------------------------------------------------------
// BiLSTM for MI455X (gfx1250, wave32): bf16 WMMA 16x16x32, fp32 accum/state,
// async-to-LDS double-buffered preactivation streaming in the recurrent scan.
// ---------------------------------------------------------------------------

#define T_LEN 512
#define BATCH 32
#define HID   512
#define G4    2048   // 4*HID
#define MROWS (T_LEN * BATCH)   // 16384

typedef __attribute__((ext_vector_type(16))) __bf16   v16bf;
typedef __attribute__((ext_vector_type(8)))  __bf16   v8bf;
typedef __attribute__((ext_vector_type(8)))  float    v8f;
typedef __attribute__((ext_vector_type(4)))  unsigned v4u;

// native f32 -> bf16 (RNE on gfx1250; lowers to v_cvt_*bf16_f32)
__device__ __forceinline__ __bf16 f2bf(float f) { return (__bf16)f; }

__device__ __forceinline__ float sigf(float x) {
  return 1.0f / (1.0f + __expf(-x));
}

// ----- WMMA 16x16x32 bf16 fragment loaders (wave32 layouts per CDNA5 ISA)
// A-frag: lane holds row m = m0+(lane&15); element e=g*8+j <-> K = k0 + g*16 + (lane>>4)*8 + j
// B-frag: lane holds col n (= row n of W, since B = W^T); same K mapping.
__device__ __forceinline__ v16bf frag_bf(const __bf16* __restrict__ row, int k0, int half) {
  struct P { v4u lo; v4u hi; } t;
  const __bf16* p = row + k0 + half * 8;
  t.lo = *(const v4u*)(p);        // K offsets +0..7   (16 bytes)
  t.hi = *(const v4u*)(p + 16);   // K offsets +16..23 (16 bytes)
  return __builtin_bit_cast(v16bf, t);
}

__device__ __forceinline__ v16bf frag_f32(const float* __restrict__ row, int k0, int half) {
  v16bf f;
  const float* p = row + k0 + half * 8;
#pragma unroll
  for (int g = 0; g < 2; ++g) {
    v8f x = *(const v8f*)(p + g * 16);
#pragma unroll
    for (int j = 0; j < 8; ++j) f[g * 8 + j] = f2bf(x[j]);
  }
  return f;
}

// ----- grid-wide barrier: monotonic counter, re-zeroed by init kernel each call
__device__ __forceinline__ void grid_sync(unsigned* ctr, unsigned target) {
  __syncthreads();
  if (threadIdx.x == 0) {
    __threadfence();
    __hip_atomic_fetch_add(ctr, 1u, __ATOMIC_RELEASE, __HIP_MEMORY_SCOPE_AGENT);
    while (__hip_atomic_load(ctr, __ATOMIC_ACQUIRE, __HIP_MEMORY_SCOPE_AGENT) < target)
      __builtin_amdgcn_s_sleep(1);
  }
  __syncthreads();
}

// ---------------------------------------------------------------------------
__global__ void init_bar_kernel(unsigned* bar) { bar[threadIdx.x] = 0u; }

// vectorized fp32 -> bf16 weight conversion (n multiple of 2048)
__global__ void cvt_bf16_kernel(const float* __restrict__ in, __bf16* __restrict__ out, int n) {
  int i = (blockIdx.x * blockDim.x + threadIdx.x) * 8;
  if (i < n) {
    v8f x = *(const v8f*)(in + i);
    v8bf o;
#pragma unroll
    for (int j = 0; j < 8; ++j) o[j] = f2bf(x[j]);
    *(v8bf*)(out + i) = o;
  }
}

// ---------------------------------------------------------------------------
// xw = X @ W^T + bias.  X: (16384 x K) fp32, W: (2048 x K) bf16, out: (16384 x 2048) fp32.
// One wave computes a 16x128 strip (8 N-tiles), A-fragment reused 8x per K-chunk.
__global__ __launch_bounds__(256)
void xw_gemm_kernel(const float* __restrict__ X, const __bf16* __restrict__ W,
                    const float* __restrict__ bias, float* __restrict__ out, int K) {
  const int NSTRIP = G4 / 128;                      // 16
  int wave = (blockIdx.x * 256 + threadIdx.x) >> 5; // 0..16383
  int mt = wave / NSTRIP;
  int ns = wave - mt * NSTRIP;
  int m0 = mt * 16;
  int n0 = ns * 128;
  int lane = threadIdx.x & 31;
  int c16 = lane & 15, half = lane >> 4;

  const float* xrow = X + (size_t)(m0 + c16) * K;
  v8f acc[8];
#pragma unroll
  for (int i = 0; i < 8; ++i) acc[i] = (v8f){0.f,0.f,0.f,0.f,0.f,0.f,0.f,0.f};

  for (int kc = 0; kc < K; kc += 32) {
    v16bf a = frag_f32(xrow, kc, half);
#pragma unroll
    for (int nt = 0; nt < 8; ++nt) {
      const __bf16* wrow = W + (size_t)(n0 + nt * 16 + c16) * K;
      v16bf b = frag_bf(wrow, kc, half);
      acc[nt] = __builtin_amdgcn_wmma_f32_16x16x32_bf16(
          false, a, false, b, (short)0, acc[nt], false, false);
    }
  }
#pragma unroll
  for (int nt = 0; nt < 8; ++nt) {
    int n = n0 + nt * 16 + c16;
    float bb = bias[n];
#pragma unroll
    for (int v = 0; v < 8; ++v) {
      int m = m0 + half * 8 + v;
      out[(size_t)m * G4 + n] = acc[nt][v] + bb;
    }
  }
}

// ---------------------------------------------------------------------------
// Persistent recurrent scan for one layer, both directions in one grid.
// 64 WGs x 256 thr: blockIdx>>5 = dir, (blockIdx&31)*16 = hidden-unit slice j0.
// Waves 0..7: (gate = w>>1) x (m-tile = w&1). One grid barrier per timestep.
// xw[t] tiles (32x64 fp32 = 8KB per WG) are streamed into LDS with
// GLOBAL_LOAD_ASYNC_TO_LDS_B128, double-buffered one step ahead.

__device__ __forceinline__ void async_xw_tile(const float* __restrict__ xw, int tq, int j0,
                                              float (*buf)[64], int tid) {
#pragma unroll
  for (int r = 0; r < 2; ++r) {
    int c = tid * 2 + r;             // 0..511 : (m, gate, quarter)
    int m = c >> 4;
    int sub = c & 15;
    int gate = sub >> 2;
    int q = sub & 3;
    const float* g = xw + ((size_t)tq * BATCH + m) * G4 + gate * HID + j0 + q * 4;
    unsigned lds = (unsigned)(uintptr_t)(&buf[m][gate * 16 + q * 4]);
    asm volatile("global_load_async_to_lds_b128 %0, %1, off"
                 :: "v"(lds), "v"(g) : "memory");
  }
}

__global__ __launch_bounds__(256)
void bilstm_scan_kernel(const float* __restrict__ xw_f, const float* __restrict__ xw_b,
                        const __bf16* __restrict__ whh_f, const __bf16* __restrict__ whh_b,
                        const int* __restrict__ lengths,
                        float* __restrict__ y,     // (T, B, 2H)
                        float* __restrict__ hn,    // (2, B, H) this layer
                        float* __restrict__ cn,    // (2, B, H) this layer
                        __bf16* __restrict__ hbf_f, __bf16* __restrict__ hbf_b,
                        unsigned* __restrict__ bar) {
  __shared__ float s_xw[2][BATCH][64];  // double-buffered xw tile (async DMA target)
  __shared__ float s_gates[BATCH][64];  // [b][gate*16 + jj]
  __shared__ float s_c[BATCH][16];
  __shared__ float s_h[BATCH][16];
  __shared__ int   s_len[BATCH];

  const int dir = blockIdx.x >> 5;
  const int j0  = (blockIdx.x & 31) * 16;
  const int tid = threadIdx.x;
  const int wv   = tid >> 5;
  const int lane = tid & 31;
  const int gate = wv >> 1;
  const int m0   = (wv & 1) * 16;
  const int c16  = lane & 15, half = lane >> 4;

  const float*  xw  = dir ? xw_b  : xw_f;
  const __bf16* whh = dir ? whh_b : whh_f;
  __bf16*       hbf = dir ? hbf_b : hbf_f;

  if (tid < BATCH) s_len[tid] = lengths[tid];

  // zero initial state (h0 = c0 = 0) for this WG's slice
#pragma unroll
  for (int r = 0; r < 2; ++r) {
    int idx = tid * 2 + r;
    int b = idx >> 4, jj = idx & 15;
    s_c[b][jj] = 0.f;
    s_h[b][jj] = 0.f;
    hbf[b * HID + j0 + jj] = f2bf(0.f);
  }

  // prime async pipeline: xw tile for t = 0 into buffer 0
  async_xw_tile(xw, dir ? (T_LEN - 1) : 0, j0, s_xw[0], tid);

  unsigned bstep = 0;
  grid_sync(bar, 64u * ++bstep);

  const int n = gate * HID + j0 + c16;                    // gate column owned by this lane
  const __bf16* wrow = whh + (size_t)n * HID;             // row n of Whh (= column n of Whh^T)
  const __bf16* arow = hbf + (size_t)(m0 + c16) * HID;    // h row for this lane

  for (int t = 0; t < T_LEN; ++t) {
    const int tq = dir ? (T_LEN - 1 - t) : t;
    const int buf = t & 1;

    // kick off next step's xw tile while the recurrent GEMM runs
    if (t + 1 < T_LEN) {
      int tq2 = dir ? (tq - 1) : (tq + 1);
      async_xw_tile(xw, tq2, j0, s_xw[buf ^ 1], tid);
    }

    // ---- phase A: gates tile = h @ Whh^T  (bf16 WMMA, fp32 accum)
    v8f acc = (v8f){0.f,0.f,0.f,0.f,0.f,0.f,0.f,0.f};
#pragma unroll
    for (int kc = 0; kc < HID; kc += 32) {
      v16bf a = frag_bf(arow, kc, half);
      v16bf b = frag_bf(wrow, kc, half);
      acc = __builtin_amdgcn_wmma_f32_16x16x32_bf16(
          false, a, false, b, (short)0, acc, false, false);
    }

    // drain this step's async tile (own wave), then WG-sync so all waves' DMA is visible
    asm volatile("s_wait_asynccnt 0x2" ::: "memory");
    __syncthreads();

#pragma unroll
    for (int v = 0; v < 8; ++v) {
      int m = m0 + half * 8 + v;
      s_gates[m][gate * 16 + c16] = acc[v] + s_xw[buf][m][gate * 16 + c16];
    }
    __syncthreads();

    // ---- phase B: element-wise cell update for this WG's 32x16 slice
#pragma unroll
    for (int r = 0; r < 2; ++r) {
      int idx = tid * 2 + r;
      int b = idx >> 4, jj = idx & 15;
      float ig = sigf (s_gates[b][jj]);
      float fg = sigf (s_gates[b][16 + jj]);
      float gg = tanhf(s_gates[b][32 + jj]);
      float og = sigf (s_gates[b][48 + jj]);
      float c_old = s_c[b][jj];
      float h_old = s_h[b][jj];
      float c_new = fg * c_old + ig * gg;
      float h_new = og * tanhf(c_new);
      bool  mk = tq < s_len[b];
      float c2 = mk ? c_new : c_old;
      float h2 = mk ? h_new : h_old;
      s_c[b][jj] = c2;
      s_h[b][jj] = h2;
      hbf[b * HID + j0 + jj] = f2bf(h2);
      y[((size_t)tq * BATCH + b) * (2 * HID) + dir * HID + j0 + jj] = mk ? h_new : 0.f;
      if (t == T_LEN - 1) {
        hn[((size_t)dir * BATCH + b) * HID + j0 + jj] = h2;
        cn[((size_t)dir * BATCH + b) * HID + j0 + jj] = c2;
      }
    }
    grid_sync(bar, 64u * ++bstep);   // publish h for next step
  }
}

// ---------------------------------------------------------------------------
extern "C" void kernel_launch(void* const* d_in, const int* in_sizes, int n_in,
                              void* d_out, int out_size, void* d_ws, size_t ws_size,
                              hipStream_t stream) {
  const float* x       = (const float*)d_in[0];
  const int*   lengths = (const int*)d_in[1];

  // Resolve Wih/Whh ordering: layer-1 Wih is 2048x1024, Whh is 2048x512.
  int wihFirst = (in_sizes[8] == G4 * (2 * HID)) ? 1 : 0;
  const float *Wih[2][2], *Whh[2][2], *Bias[2][2];
  for (int l = 0; l < 2; ++l)
    for (int d = 0; d < 2; ++d) {
      int base = 2 + (l * 2 + d) * 3;
      Wih[l][d]  = (const float*)d_in[base + (wihFirst ? 0 : 1)];
      Whh[l][d]  = (const float*)d_in[base + (wihFirst ? 1 : 0)];
      Bias[l][d] = (const float*)d_in[base + 2];
    }

  // ---- workspace layout
  char* ws = (char*)d_ws;
  size_t off = 0;
  unsigned* bar = (unsigned*)(ws + off); off += 256;     // 2 barrier counters (cache-line apart)
  __bf16* whh_bf[2][2];
  for (int l = 0; l < 2; ++l)
    for (int d = 0; d < 2; ++d) { whh_bf[l][d] = (__bf16*)(ws + off); off += (size_t)G4 * HID * 2; }
  __bf16* wih_bf[2][2];
  int inK[2] = {HID, 2 * HID};
  for (int l = 0; l < 2; ++l)
    for (int d = 0; d < 2; ++d) { wih_bf[l][d] = (__bf16*)(ws + off); off += (size_t)G4 * inK[l] * 2; }
  __bf16* hbf[2];
  for (int d = 0; d < 2; ++d) { hbf[d] = (__bf16*)(ws + off); off += (size_t)BATCH * HID * 2; }
  off = (off + 255) & ~(size_t)255;
  float* xwbuf[2];
  for (int d = 0; d < 2; ++d) { xwbuf[d] = (float*)(ws + off); off += (size_t)MROWS * G4 * 4; }
  float* y0 = (float*)(ws + off); off += (size_t)MROWS * (2 * HID) * 4;

  float* out_x  = (float*)d_out;
  float* out_hn = out_x  + (size_t)MROWS * (2 * HID);
  float* out_cn = out_hn + (size_t)4 * BATCH * HID;

  // ---- barrier init (re-run every call / graph replay)
  init_bar_kernel<<<1, 64, 0, stream>>>(bar);

  // ---- weight conversion fp32 -> bf16 (8 elements / thread)
  for (int l = 0; l < 2; ++l)
    for (int d = 0; d < 2; ++d) {
      int nh = G4 * HID;
      cvt_bf16_kernel<<<nh / (256 * 8), 256, 0, stream>>>(Whh[l][d], whh_bf[l][d], nh);
      int ni = G4 * inK[l];
      cvt_bf16_kernel<<<ni / (256 * 8), 256, 0, stream>>>(Wih[l][d], wih_bf[l][d], ni);
    }

  // ---- layers
  const float* xin = x;
  for (int l = 0; l < 2; ++l) {
    int K = inK[l];
    for (int d = 0; d < 2; ++d)
      xw_gemm_kernel<<<(MROWS / 16) * (G4 / 128) / 8, 256, 0, stream>>>(
          xin, wih_bf[l][d], Bias[l][d], xwbuf[d], K);

    float* yout = (l == 1) ? out_x : y0;
    bilstm_scan_kernel<<<64, 256, 0, stream>>>(
        xwbuf[0], xwbuf[1], whh_bf[l][0], whh_bf[l][1], lengths,
        yout,
        out_hn + (size_t)l * 2 * BATCH * HID,
        out_cn + (size_t)l * 2 * BATCH * HID,
        hbf[0], hbf[1], bar + 32 * l);
    xin = y0;
  }
}
